// ReformerPP_10926396801632
// MI455X (gfx1250) — matrix operand order repaired and checked by reading, not compiled, not run on hardware
//
#include <hip/hip_runtime.h>
#include <math.h>
#include <stdint.h>

// ---------------- problem constants ----------------
constexpr int B_ = 2, S_ = 4096, T_ = 2048, D_ = 512, H_ = 8, L_ = 2;
constexpr int DH_ = 64, BUCKET_ = 64, NH_ = 4, NC_ = S_ / BUCKET_, NB2_ = 32;
constexpr int V_ = 32000;

typedef __bf16 bf16_t;
typedef __attribute__((ext_vector_type(16))) __bf16 v16bf;
typedef __attribute__((ext_vector_type(8)))  float  v8f;
typedef __attribute__((ext_vector_type(4)))  unsigned int u32x4;
typedef __attribute__((ext_vector_type(8)))  int          i32x8;
typedef __attribute__((ext_vector_type(4)))  int          i32x4;

// ---------------- WMMA fragment loaders (LDS, row-major [row][k]) ----------
// A-matrix 16x32 bf16: lanes 0-15 -> K {0..7,16..23}; lanes 16-31 -> K {8..15,24..31}
__device__ inline v16bf load_frag_A(const bf16_t* rowBase, int lane) {
  const unsigned int* r = (const unsigned int*)rowBase;
  int kb2 = (lane & 16) ? 4 : 0;
  union { v16bf v; unsigned int u[8]; } f;
#pragma unroll
  for (int i = 0; i < 4; ++i) { f.u[i] = r[kb2 + i]; f.u[4 + i] = r[kb2 + 8 + i]; }
  return f.v;
}
// B-matrix 32x16 bf16 stored as Bs[n][k]: lanes 0-15 -> K {0..7,16..23}; lanes 16-31 -> K {8..15,24..31}
__device__ inline v16bf load_frag_B(const bf16_t* rowBase, int lane) {
  const unsigned int* r = (const unsigned int*)rowBase;
  int kb2 = (lane & 16) ? 4 : 0;
  union { v16bf v; unsigned int u[8]; } f;
#pragma unroll
  for (int i = 0; i < 4; ++i) { f.u[i] = r[kb2 + i]; f.u[4 + i] = r[kb2 + 8 + i]; }
  return f.v;
}

// ---------------- TDM: DMA a 64 x 256 bf16 panel (row stride K) into LDS ---
__device__ inline void tdm_load_panel(const bf16_t* gsrc, bf16_t* lds_dst, int K) {
  unsigned long long gaddr = (unsigned long long)(uintptr_t)gsrc;
  unsigned laddr = (unsigned)(unsigned long long)(uintptr_t)lds_dst; // LDS byte offset
  u32x4 g0;
  g0[0] = 1u;                                               // count=1 (valid)
  g0[1] = laddr;                                            // lds_addr
  g0[2] = (unsigned)(gaddr & 0xFFFFFFFFull);                // global_addr[31:0]
  g0[3] = (unsigned)((gaddr >> 32) & 0x1FFFFFFu) | (2u << 30); // addr[56:32] | type=2
  i32x8 g1;
  g1[0] = (int)(1u << 16);      // workgroup_mask=0, data_size=1 (2 bytes)
  g1[1] = (int)(256u << 16);    // tensor_dim0[15:0]=256
  g1[2] = (int)(64u << 16);     // tensor_dim0 hi=0 | tensor_dim1[15:0]=64
  g1[3] = (int)(256u << 16);    // tensor_dim1 hi=0 | tile_dim0=256
  g1[4] = (int)64;              // tile_dim1=64, tile_dim2=0
  g1[5] = (int)K;               // tensor_dim0_stride[31:0] = K elements
  g1[6] = 0;                    // stride hi, tensor_dim1_stride lo
  g1[7] = 0;
  i32x4 z4 = {0, 0, 0, 0};
  i32x8 z8 = {0, 0, 0, 0, 0, 0, 0, 0};
  __builtin_amdgcn_tensor_load_to_lds(g0, g1, z4, z4, z8, 0); // 6-arg (clang-23) form
  __builtin_amdgcn_s_wait_tensorcnt(0);
}

// ---------------- generic bf16 WMMA GEMM: C[M][N] = A[M][K] * Bt[N][K]^T ----
// Requires: M%128==0, N%64==0, K%256==0 (here K==512 for every call).
template <bool BIAS, bool RELU>
__global__ __launch_bounds__(256) void gemm_bf16_kernel(
    const bf16_t* __restrict__ A, const bf16_t* __restrict__ Bt,
    const float* __restrict__ bias, float* __restrict__ C,
    int M, int N, int K) {
  __shared__ bf16_t Bp[64][256];     // weight panel (TDM resident), 32 KB
  __shared__ bf16_t As[2][128][32];  // double-buffered activation tile, 16 KB
  const int t = threadIdx.x, wave = t >> 5, lane = t & 31;
  const int m0 = blockIdx.y * 128, n0 = blockIdx.x * 64;
  const int rowA = t >> 1, halfA = t & 1;
  v8f acc[4] = {};
  for (int kh = 0; kh < K; kh += 256) {
    __syncthreads();  // prior readers of Bp / As done
    if (wave == 0)    // DMA 64 x 256 weight panel into LDS via Tensor Data Mover
      tdm_load_panel(Bt + (size_t)n0 * K + kh, &Bp[0][0], K);
    // preload A tile 0 of this K-half
    uint4 ra0, ra1;
    {
      const uint4* g = (const uint4*)(A + (size_t)(m0 + rowA) * K + kh + halfA * 16);
      ra0 = g[0]; ra1 = g[1];
    }
    {
      uint4* sp = (uint4*)(&As[0][rowA][halfA * 16]);
      sp[0] = ra0; sp[1] = ra1;
    }
    __syncthreads();
#pragma unroll
    for (int s = 0; s < 8; ++s) {
      if (s < 7) { // software pipeline: fetch next A tile while computing
        const uint4* g =
            (const uint4*)(A + (size_t)(m0 + rowA) * K + kh + (s + 1) * 32 + halfA * 16);
        ra0 = g[0]; ra1 = g[1];
      }
      v16bf af = load_frag_A(&As[s & 1][wave * 16 + (lane & 15)][0], lane);
#pragma unroll
      for (int nt = 0; nt < 4; ++nt) {
        v16bf bfr = load_frag_B(&Bp[nt * 16 + (lane & 15)][s * 32], lane);
        acc[nt] = __builtin_amdgcn_wmma_f32_16x16x32_bf16(
            false, af, false, bfr, (short)0, acc[nt], false, false);
      }
      if (s < 7) {
        uint4* sp = (uint4*)(&As[(s + 1) & 1][rowA][halfA * 16]);
        sp[0] = ra0; sp[1] = ra1;
        __syncthreads();
      }
    }
  }
  const int mloc = wave * 16 + ((lane & 16) ? 8 : 0);
  const int ncol = lane & 15;
#pragma unroll
  for (int nt = 0; nt < 4; ++nt) {
    int n = n0 + nt * 16 + ncol;
    float bv = 0.f;
    if constexpr (BIAS) bv = bias[n];
#pragma unroll
    for (int r = 0; r < 8; ++r) {
      float val = acc[nt][r] + bv;
      if constexpr (RELU) val = fmaxf(val, 0.f);
      C[(size_t)(m0 + mloc + r) * N + n] = val;
    }
  }
}

// ---------------- small elementwise / layout kernels ----------------------
__global__ void embed_kernel(const int* __restrict__ src, const float* __restrict__ emb,
                             const float* __restrict__ pos, float* __restrict__ x1) {
  size_t idx = (size_t)blockIdx.x * 256 + threadIdx.x;
  if (idx >= (size_t)B_ * S_ * D_) return;
  int d = (int)(idx % D_);
  size_t bs = idx / D_;
  int s = (int)(bs % S_);
  x1[idx] = emb[(size_t)src[bs] * D_ + d] + pos[(size_t)s * D_ + d];
}

__global__ void cast_bf16_kernel(const float* __restrict__ x, bf16_t* __restrict__ y, size_t n) {
  size_t i = (size_t)blockIdx.x * 256 + threadIdx.x;
  if (i < n) y[i] = (bf16_t)x[i];
}

__global__ void transpose_cast_kernel(const float* __restrict__ W, bf16_t* __restrict__ Wt,
                                      int K, int N) {
  size_t idx = (size_t)blockIdx.x * 256 + threadIdx.x;
  if (idx >= (size_t)N * K) return;
  int k = (int)(idx % K);
  int n = (int)(idx / K);
  Wt[idx] = (bf16_t)W[(size_t)k * N + n];
}

__global__ __launch_bounds__(256) void ln_kernel(const float* __restrict__ x,
                                                 const float* __restrict__ sc,
                                                 const float* __restrict__ bi,
                                                 float* __restrict__ y) {
  __shared__ float red[256];
  const int row = blockIdx.x, t = threadIdx.x;
  const float* xr = x + (size_t)row * D_;
  float a = xr[t], b = xr[t + 256];
  red[t] = a + b;
  __syncthreads();
  for (int o = 128; o > 0; o >>= 1) { if (t < o) red[t] += red[t + o]; __syncthreads(); }
  float mean = red[0] * (1.0f / D_);
  __syncthreads();
  float d0 = a - mean, d1 = b - mean;
  red[t] = d0 * d0 + d1 * d1;
  __syncthreads();
  for (int o = 128; o > 0; o >>= 1) { if (t < o) red[t] += red[t + o]; __syncthreads(); }
  float r = rsqrtf(red[0] * (1.0f / D_) + 1e-5f);
  y[(size_t)row * D_ + t]       = d0 * r * sc[t] + bi[t];
  y[(size_t)row * D_ + t + 256] = d1 * r * sc[t + 256] + bi[t + 256];
}

__global__ void resid_kernel(float* __restrict__ x, const float* __restrict__ a,
                             const float* __restrict__ g) {
  size_t idx = (size_t)blockIdx.x * 256 + threadIdx.x;
  if (idx >= (size_t)B_ * S_ * D_) return;
  int d = (int)(idx % D_);
  float gv = 1.f / (1.f + expf(-g[d]));
  x[idx] += gv * a[idx];
}

__global__ void slice_avg_kernel(const float* __restrict__ x1, const float* __restrict__ x2,
                                 float* __restrict__ o) {
  size_t idx = (size_t)blockIdx.x * 256 + threadIdx.x;
  if (idx >= (size_t)B_ * T_ * D_) return;
  int d = (int)(idx % D_);
  size_t bt = idx / D_;
  int tt = (int)(bt % T_);
  int b  = (int)(bt / T_);
  size_t si = ((size_t)b * S_ + tt) * D_ + d;
  o[idx] = 0.5f * (x1[si] + x2[si]);
}

// ---------------- LSH hashing + bitonic sort ------------------------------
__global__ __launch_bounds__(256) void lsh_hash_kernel(const float* __restrict__ qk,
                                                       const float* __restrict__ R,
                                                       unsigned int* __restrict__ keys) {
  __shared__ float Rs[DH_ * NB2_];
  const int h = blockIdx.y, b = blockIdx.z, t = threadIdx.x;
  for (int i = t; i < DH_ * NB2_; i += 256) Rs[i] = R[(size_t)h * DH_ * NB2_ + i];
  __syncthreads();
  const int s = blockIdx.x * 256 + t;
  const float* q = qk + ((size_t)b * S_ + s) * D_ + h * DH_;
  float qr[DH_];
#pragma unroll
  for (int d = 0; d < DH_; ++d) qr[d] = q[d];
  float best = -3.4e38f; int bb = 0;
  for (int n = 0; n < NB2_; ++n) {
    float p = 0.f;
#pragma unroll
    for (int d = 0; d < DH_; ++d) p += qr[d] * Rs[d * NB2_ + n];
    if (p > best)  { best = p;  bb = n; }
    if (-p > best) { best = -p; bb = n + NB2_; }
  }
  keys[((size_t)b * H_ + h) * S_ + s] = (unsigned)(bb * S_ + s);
}

__global__ __launch_bounds__(1024) void bitonic_sort_kernel(const unsigned int* __restrict__ keys,
                                                            int* __restrict__ perm) {
  __shared__ unsigned int k[S_];
  const int bh = blockIdx.x, t = threadIdx.x;
  const unsigned int* gk = keys + (size_t)bh * S_;
  for (int i = t; i < S_; i += 1024) k[i] = gk[i];
  __syncthreads();
  for (int len = 2; len <= S_; len <<= 1) {
    for (int j = len >> 1; j > 0; j >>= 1) {
      for (int i = t; i < S_; i += 1024) {
        int ixj = i ^ j;
        if (ixj > i) {
          unsigned a = k[i], c = k[ixj];
          bool up = ((i & len) == 0);
          if ((a > c) == up) { k[i] = c; k[ixj] = a; }
        }
      }
      __syncthreads();
    }
  }
  int* gp = perm + (size_t)bh * S_;
  for (int i = t; i < S_; i += 1024) gp[i] = (int)(k[i] & (unsigned)(S_ - 1));
}

// ---------------- LSH chunked attention (WMMA x2 + wave softmax) ----------
__global__ __launch_bounds__(128) void lsh_chunk_attn_kernel(
    const float* __restrict__ qk, const float* __restrict__ v,
    const int* __restrict__ perm, float* __restrict__ g_out, int accumulate) {
  __shared__ bf16_t Qs[64][64];
  __shared__ bf16_t Ks[128][64];
  __shared__ bf16_t Vt[64][128];   // Vt[d][j]
  __shared__ bf16_t Pb[64][128];
  __shared__ int prm[128];
  const int c = blockIdx.x, h = blockIdx.y, b = blockIdx.z;
  const int t = threadIdx.x, wave = t >> 5, lane = t & 31;
  const int cprev = (c + NC_ - 1) % NC_;
  const int* pbase = perm + ((size_t)b * H_ + h) * S_;
  if (t < 64) prm[t] = pbase[c * BUCKET_ + t];
  else        prm[t] = pbase[cprev * BUCKET_ + (t - 64)];
  __syncthreads();
  { // gather sorted rows (row j of keys / values)
    int j = t;
    int orig = prm[j];
    const float* qrow = qk + ((size_t)b * S_ + orig) * D_ + h * DH_;
    const float* vrow = v  + ((size_t)b * S_ + orig) * D_ + h * DH_;
#pragma unroll 4
    for (int d = 0; d < DH_; ++d) {
      bf16_t qb = (bf16_t)qrow[d];
      Ks[j][d] = qb;
      if (j < 64) Qs[j][d] = qb;
      Vt[d][j] = (bf16_t)vrow[d];
    }
  }
  __syncthreads();
  // scores: wave owns rows [wave*16, wave*16+16), 8 N-tiles of 16, K=64
  v8f sacc[8] = {};
#pragma unroll
  for (int kk = 0; kk < 2; ++kk) {
    v16bf af = load_frag_A(&Qs[wave * 16 + (lane & 15)][kk * 32], lane);
#pragma unroll
    for (int nt = 0; nt < 8; ++nt) {
      v16bf bfr = load_frag_B(&Ks[nt * 16 + (lane & 15)][kk * 32], lane);
      sacc[nt] = __builtin_amdgcn_wmma_f32_16x16x32_bf16(
          false, af, false, bfr, (short)0, sacc[nt], false, false);
    }
  }
  // in-register softmax per row (16-lane groups share a row)
  const float scale = 0.125f; // DH^-0.5
#pragma unroll
  for (int r = 0; r < 8; ++r) {
    int row = wave * 16 + ((lane & 16) ? 8 : 0) + r;
    float sv[8]; float mx = -3.4e38f;
#pragma unroll
    for (int nt = 0; nt < 8; ++nt) { sv[nt] = sacc[nt][r] * scale; mx = fmaxf(mx, sv[nt]); }
    for (int off = 1; off < 16; off <<= 1) mx = fmaxf(mx, __shfl_xor(mx, off, 16));
    float sum = 0.f;
#pragma unroll
    for (int nt = 0; nt < 8; ++nt) { sv[nt] = __expf(sv[nt] - mx); sum += sv[nt]; }
    for (int off = 1; off < 16; off <<= 1) sum += __shfl_xor(sum, off, 16);
    float inv = 1.0f / sum;
#pragma unroll
    for (int nt = 0; nt < 8; ++nt) Pb[row][nt * 16 + (lane & 15)] = (bf16_t)(sv[nt] * inv);
  }
  __syncthreads();
  // out = P(64x128) @ V(128x64): 4 accumulators, K loop of 4
  v8f oacc[4] = {};
#pragma unroll
  for (int kk = 0; kk < 4; ++kk) {
    v16bf af = load_frag_A(&Pb[wave * 16 + (lane & 15)][kk * 32], lane);
#pragma unroll
    for (int nt = 0; nt < 4; ++nt) {
      v16bf bfr = load_frag_B(&Vt[nt * 16 + (lane & 15)][kk * 32], lane);
      oacc[nt] = __builtin_amdgcn_wmma_f32_16x16x32_bf16(
          false, af, false, bfr, (short)0, oacc[nt], false, false);
    }
  }
  // scatter back via perm (one writer per (orig,d) within a round)
  const int rbase = wave * 16 + ((lane & 16) ? 8 : 0);
#pragma unroll
  for (int r = 0; r < 8; ++r) {
    int i = rbase + r;
    int orig = prm[i];
    float* gr = g_out + ((size_t)b * S_ + orig) * D_ + h * DH_;
#pragma unroll
    for (int nt = 0; nt < 4; ++nt) {
      int d = nt * 16 + (lane & 15);
      float val = oacc[nt][r] * 0.25f; // 1/N_HASHES
      if (accumulate) gr[d] += val; else gr[d] = val;
    }
  }
}

// ---------------- local window attention ----------------------------------
__global__ __launch_bounds__(256) void local_attn_kernel(const float* __restrict__ qk,
                                                         const float* __restrict__ v,
                                                         float* __restrict__ l_out) {
  const int h = blockIdx.y, b = blockIdx.z;
  const int s = blockIdx.x * 256 + threadIdx.x;
  const float* q = qk + ((size_t)b * S_ + s) * D_ + h * DH_;
  float qr[DH_];
#pragma unroll
  for (int d = 0; d < DH_; ++d) qr[d] = q[d];
  float sc[9]; float mx = -3.4e38f;
#pragma unroll
  for (int w = 0; w < 9; ++w) {
    int off = w - 4, sp = s + off;
    bool valid = (sp >= 0) && (sp < S_);
    int spw = (sp + S_) & (S_ - 1);
    const float* kr = qk + ((size_t)b * S_ + spw) * D_ + h * DH_;
    float acc = 0.f;
#pragma unroll
    for (int d = 0; d < DH_; ++d) acc += qr[d] * kr[d];
    sc[w] = valid ? acc * 0.125f : -1e9f;
    mx = fmaxf(mx, sc[w]);
  }
  float sum = 0.f;
#pragma unroll
  for (int w = 0; w < 9; ++w) { sc[w] = __expf(sc[w] - mx); sum += sc[w]; }
  float inv = 1.f / sum;
  float out[DH_];
#pragma unroll
  for (int d = 0; d < DH_; ++d) out[d] = 0.f;
#pragma unroll
  for (int w = 0; w < 9; ++w) {
    int spw = (s + w - 4 + S_) & (S_ - 1);
    const float* vr = v + ((size_t)b * S_ + spw) * D_ + h * DH_;
    float p = sc[w] * inv;
#pragma unroll
    for (int d = 0; d < DH_; ++d) out[d] += p * vr[d];
  }
  float* lo = l_out + ((size_t)b * S_ + s) * D_ + h * DH_;
#pragma unroll
  for (int d = 0; d < DH_; ++d) lo[d] = out[d];
}

// ---------------- gate + combine (+ reg accumulation) ---------------------
__global__ __launch_bounds__(256) void combine_kernel(const float* __restrict__ qk,
                                                      const float* __restrict__ g_out,
                                                      const float* __restrict__ l_out,
                                                      const float* __restrict__ gw,
                                                      float* __restrict__ o,
                                                      float* __restrict__ reg_sum) {
  __shared__ float gwS[DH_];
  __shared__ float red[256];
  const int h = blockIdx.y, b = blockIdx.z, t = threadIdx.x;
  const int s = blockIdx.x * 256 + t;
  if (t < DH_) gwS[t] = gw[h * DH_ + t];
  __syncthreads();
  const size_t base = ((size_t)b * S_ + s) * D_ + h * DH_;
  const float* q = qk + base;
  float dot = 0.f;
#pragma unroll
  for (int d = 0; d < DH_; ++d) dot += q[d] * gwS[d];
  float gate = 1.f / (1.f + expf(-dot));
  const float* gr = g_out + base;
  const float* lr = l_out + base;
  float* orow = o + base;
#pragma unroll
  for (int d = 0; d < DH_; ++d) orow[d] = gate * gr[d] + (1.f - gate) * lr[d];
  red[t] = gate * (1.f - gate);
  __syncthreads();
  for (int off = 128; off > 0; off >>= 1) { if (t < off) red[t] += red[t + off]; __syncthreads(); }
  if (t == 0) atomicAdd(reg_sum, red[0]);
}

__global__ void finalize_reg_kernel(const float* __restrict__ reg_sum, float* __restrict__ out_reg) {
  out_reg[0] = reg_sum[0] / (float)((size_t)B_ * H_ * S_);
}

// ---------------- host-side orchestration ---------------------------------
extern "C" void kernel_launch(void* const* d_in, const int* in_sizes, int n_in,
                              void* d_out, int out_size, void* d_ws, size_t ws_size,
                              hipStream_t stream) {
  (void)in_sizes; (void)n_in; (void)out_size; (void)ws_size;
  const int*   src       = (const int*)d_in[0];
  const float* emb       = (const float*)d_in[2];
  const float* pos       = (const float*)d_in[3];
  const float* attn_ln_s = (const float*)d_in[4];
  const float* attn_ln_b = (const float*)d_in[5];
  const float* Wqk       = (const float*)d_in[6];
  const float* Wv        = (const float*)d_in[7];
  const float* Wo        = (const float*)d_in[8];
  const float* hashR     = (const float*)d_in[9];
  const float* gate_w    = (const float*)d_in[10];
  const float* rev_g1    = (const float*)d_in[11];
  const float* rev_g2    = (const float*)d_in[12];
  const float* ffn_ln_s  = (const float*)d_in[13];
  const float* ffn_ln_b  = (const float*)d_in[14];
  const float* ffn_W1    = (const float*)d_in[15];
  const float* ffn_b1    = (const float*)d_in[16];
  const float* ffn_W2    = (const float*)d_in[17];
  const float* ffn_b2    = (const float*)d_in[18];
  const float* out_W     = (const float*)d_in[19];
  const float* out_b     = (const float*)d_in[20];
  float* out = (float*)d_out;

  const size_t NSD = (size_t)B_ * S_ * D_;   // 4,194,304
  const size_t NTD = (size_t)B_ * T_ * D_;   // 2,097,152

  char* p = (char*)d_ws;
  auto carve = [&](size_t bytes) { char* r = p; p += (bytes + 255) & ~(size_t)255; return r; };
  float*    x1     = (float*)carve(NSD * 4);
  float*    x2     = (float*)carve(NSD * 4);
  float*    bA     = (float*)carve(NSD * 4);   // xn / ocomb / h / out-slice
  float*    bB     = (float*)carve(NSD * 4);   // qk / f1
  float*    bC     = (float*)carve(NSD * 4);   // v  / f2
  float*    bD     = (float*)carve(NSD * 4);   // g_out / a_out
  float*    bE     = (float*)carve(NSD * 4);   // l_out
  bf16_t*   act_bf = (bf16_t*)carve(NSD * 2);
  bf16_t*   wts    = (bf16_t*)carve((size_t)D_ * D_ * 2);
  bf16_t*   wtb    = (bf16_t*)carve((size_t)V_ * D_ * 2);
  unsigned* keys   = (unsigned*)carve((size_t)B_ * H_ * S_ * 4);
  int*      perm   = (int*)carve((size_t)B_ * H_ * S_ * 4);
  float*    regsum = (float*)carve(256);

  (void)hipMemsetAsync(x2, 0, NSD * 4, stream);
  (void)hipMemsetAsync(regsum, 0, 4, stream);

  const dim3 blk256(256);
  const int gNSD = (int)(NSD / 256);
  const int gNTD = (int)(NTD / 256);
  const dim3 gBHS(S_ / 256, H_, B_);
  const dim3 gChunk(NC_, H_, B_);

  auto castf = [&](const float* x, bf16_t* y, size_t n) {
    cast_bf16_kernel<<<(unsigned)((n + 255) / 256), blk256, 0, stream>>>(x, y, n);
  };
  auto transW = [&](const float* W, bf16_t* Wt, int K, int N) {
    size_t n = (size_t)N * K;
    transpose_cast_kernel<<<(unsigned)((n + 255) / 256), blk256, 0, stream>>>(W, Wt, K, N);
  };

  embed_kernel<<<gNSD, blk256, 0, stream>>>(src, emb, pos, x1);

  for (int l = 0; l < L_; ++l) {
    const float* Wqk_l = Wqk + (size_t)l * D_ * D_;
    const float* Wv_l  = Wv  + (size_t)l * D_ * D_;
    const float* Wo_l  = Wo  + (size_t)l * D_ * D_;
    const float* W1_l  = ffn_W1 + (size_t)l * D_ * D_;
    const float* W2_l  = ffn_W2 + (size_t)l * D_ * D_;

    // attention LN on x2
    ln_kernel<<<B_ * S_, blk256, 0, stream>>>(x2, attn_ln_s + l * D_, attn_ln_b + l * D_, bA);
    castf(bA, act_bf, NSD);

    // qk = xn @ Wqk ; v = xn @ Wv   (WMMA + TDM weight panels)
    dim3 g1(D_ / 64, (B_ * S_) / 128);
    transW(Wqk_l, wts, D_, D_);
    gemm_bf16_kernel<false, false><<<g1, blk256, 0, stream>>>(act_bf, wts, nullptr, bB, B_ * S_, D_, D_);
    transW(Wv_l, wts, D_, D_);
    gemm_bf16_kernel<false, false><<<g1, blk256, 0, stream>>>(act_bf, wts, nullptr, bC, B_ * S_, D_, D_);

    // LSH attention, 4 hash rounds
    for (int r = 0; r < NH_; ++r) {
      const float* R = hashR + ((size_t)(l * NH_ + r)) * H_ * DH_ * NB2_;
      lsh_hash_kernel<<<gBHS, blk256, 0, stream>>>(bB, R, keys);
      bitonic_sort_kernel<<<B_ * H_, 1024, 0, stream>>>(keys, perm);
      lsh_chunk_attn_kernel<<<gChunk, 128, 0, stream>>>(bB, bC, perm, bD, (r > 0) ? 1 : 0);
    }

    // local window attention + gated combine
    local_attn_kernel<<<gBHS, blk256, 0, stream>>>(bB, bC, bE);
    combine_kernel<<<gBHS, blk256, 0, stream>>>(bB, bD, bE, gate_w + (size_t)l * H_ * DH_, bA, regsum);

    // a_out = combined @ Wo
    castf(bA, act_bf, NSD);
    transW(Wo_l, wts, D_, D_);
    gemm_bf16_kernel<false, false><<<g1, blk256, 0, stream>>>(act_bf, wts, nullptr, bD, B_ * S_, D_, D_);

    // y1 = x1 + sigmoid(rev_g1)*a_out (in place)
    resid_kernel<<<gNSD, blk256, 0, stream>>>(x1, bD, rev_g1 + (size_t)l * D_);

    // FFN
    ln_kernel<<<B_ * S_, blk256, 0, stream>>>(x1, ffn_ln_s + l * D_, ffn_ln_b + l * D_, bA);
    castf(bA, act_bf, NSD);
    transW(W1_l, wts, D_, D_);
    gemm_bf16_kernel<true, true><<<g1, blk256, 0, stream>>>(act_bf, wts, ffn_b1 + (size_t)l * D_, bB, B_ * S_, D_, D_);
    castf(bB, act_bf, NSD);
    transW(W2_l, wts, D_, D_);
    gemm_bf16_kernel<true, false><<<g1, blk256, 0, stream>>>(act_bf, wts, ffn_b2 + (size_t)l * D_, bC, B_ * S_, D_, D_);

    // y2 = x2 + sigmoid(rev_g2)*f (in place)
    resid_kernel<<<gNSD, blk256, 0, stream>>>(x2, bC, rev_g2 + (size_t)l * D_);
  }

  // out = (x1+x2)/2 sliced to T, then vocab projection (dominant GEMM)
  slice_avg_kernel<<<gNTD, blk256, 0, stream>>>(x1, x2, bA);
  castf(bA, act_bf, NTD);
  transW(out_W, wtb, D_, V_);
  dim3 g2(V_ / 64, (B_ * T_) / 128);
  gemm_bf16_kernel<true, false><<<g2, blk256, 0, stream>>>(act_bf, wtb, out_b, out, B_ * T_, V_, D_);

  finalize_reg_kernel<<<1, 1, 0, stream>>>(regsum, out + (size_t)B_ * T_ * V_);
}
